// BiaffineSpan_59098749993537
// MI455X (gfx1250) — matrix operand order repaired and checked by reading, not compiled, not run on hardware
//
#include <hip/hip_runtime.h>

// ---------------------------------------------------------------------------
// BiaffineSpan for MI455X (gfx1250, wave32, WMMA).
// Hs/He via f32 LDS-tiled GEMMs; bilinear einsums via v_wmma_f32_16x16x32_bf16
// with bf16 hi/lo precision splitting (3 WMMAs per 16x16x32 step).
// ---------------------------------------------------------------------------

typedef __attribute__((ext_vector_type(16))) __bf16       v16bf;
typedef __attribute__((ext_vector_type(8)))  float        v8f;
typedef __attribute__((ext_vector_type(4)))  unsigned int u32x4;
typedef __attribute__((ext_vector_type(4)))  float        f32x4;

union FragU { u32x4 u[2]; v16bf v; };

static __device__ __forceinline__ v8f wmma_acc(v16bf a, v16bf b, v8f c) {
    return __builtin_amdgcn_wmma_f32_16x16x32_bf16(false, a, false, b, (short)0, c,
                                                   false, false);
}

// A fragment (16x32, M x K): lane = M row (lane&15), two contiguous 8-elem K
// chunks at k-offset (lane<16 ? 0 : 8) and +16.  Source is row-major [M][K].
static __device__ __forceinline__ v16bf load_afrag(const __bf16* base, int stride, int lane) {
    const int m  = lane & 15;
    const int kb = (lane < 16) ? 0 : 8;
    const __bf16* p = base + (size_t)m * stride + kb;
    FragU f;
    f.u[0] = *(const u32x4*)(p);
    f.u[1] = *(const u32x4*)(p + 16);
    return f.v;
}

// B fragment (32x16, K x N): lane = N column (lane&15), two contiguous 8-elem K
// chunks at k-offset (lane<16 ? 0 : 8) and +16.  Source is N-major [N][K] with given stride.
static __device__ __forceinline__ v16bf load_bfrag(const __bf16* base, int stride, int lane) {
    const int n  = lane & 15;
    const int kg = (lane < 16) ? 0 : 8;
    const __bf16* p = base + (size_t)n * stride + kg;
    FragU f;
    f.u[0] = *(const u32x4*)(p);
    f.u[1] = *(const u32x4*)(p + 16);
    return f.v;
}

// ---------------------------------------------------------------------------
// f32 GEMM: Y[m][n] = act( sum_k X[m][k] * W[n*wstride + woff + k] + bias[n] )
// block = 256 threads, 64x64 tile, 4x4 per thread.  Used for the MLPs & Ls/Le.
// ---------------------------------------------------------------------------
__global__ void __launch_bounds__(256) gemm_xwt_kernel(
    const float* __restrict__ X, const float* __restrict__ W,
    const float* __restrict__ bias, float* __restrict__ Y,
    int M, int N, int K, int wstride, int woff, int relu)
{
    __shared__ float Xs[16][65];
    __shared__ float Ws[16][65];
    const int tid = threadIdx.x;
    const int tx = tid & 15, ty = tid >> 4;
    const int row0 = blockIdx.x * 64, col0 = blockIdx.y * 64;
    float acc[4][4] = {};
    for (int k0 = 0; k0 < K; k0 += 16) {
        __syncthreads();
        #pragma unroll
        for (int r = 0; r < 4; ++r) {
            int idx = tid + 256 * r;
            int kk = idx & 15, mm = idx >> 4;
            Xs[kk][mm] = X[(size_t)(row0 + mm) * K + k0 + kk];
            Ws[kk][mm] = W[(size_t)(col0 + mm) * wstride + woff + k0 + kk];
        }
        __syncthreads();
        #pragma unroll
        for (int kk = 0; kk < 16; ++kk) {
            float xa[4], wb[4];
            #pragma unroll
            for (int a = 0; a < 4; ++a) xa[a] = Xs[kk][ty * 4 + a];
            #pragma unroll
            for (int b = 0; b < 4; ++b) wb[b] = Ws[kk][tx * 4 + b];
            #pragma unroll
            for (int a = 0; a < 4; ++a)
                #pragma unroll
                for (int b = 0; b < 4; ++b) acc[a][b] = fmaf(xa[a], wb[b], acc[a][b]);
        }
    }
    #pragma unroll
    for (int a = 0; a < 4; ++a)
        #pragma unroll
        for (int b = 0; b < 4; ++b) {
            int n = col0 + tx * 4 + b;
            float v = acc[a][b] + (bias ? bias[n] : 0.0f);
            if (relu) v = fmaxf(v, 0.0f);
            Y[(size_t)(row0 + ty * 4 + a) * N + n] = v;
        }
}

// f32 -> bf16 hi/lo planes (x = hi + lo)
__global__ void __launch_bounds__(256) split_kernel(
    const float* __restrict__ src, __bf16* __restrict__ hi, __bf16* __restrict__ lo, int n)
{
    int i = blockIdx.x * 256 + threadIdx.x;
    if (i < n) {
        float x = src[i];
        __bf16 h = (__bf16)x;
        hi[i] = h;
        lo[i] = (__bf16)(x - (float)h);
    }
}

// ---------------------------------------------------------------------------
// Pass 1: T[b][cl][i][e] = sum_d Hs[b,i,d] * W1[c,d,e], stored as bf16 hi/lo.
// grid = (32 i-chunks of 32 rows over flattened b*512+i, NCH channels).
// W1 K-chunk (32 x 768) is staged to LDS transposed ([e][k], N-major) with
// f32->bf16 conversion; hi plane then lo plane reuse the same LDS buffer.
// ---------------------------------------------------------------------------
__global__ void __launch_bounds__(256) pass1_T_kernel(
    const __bf16* __restrict__ Hshi, const __bf16* __restrict__ Hslo,
    const float*  __restrict__ W1,
    __bf16* __restrict__ Thi, __bf16* __restrict__ Tlo,
    int NCH, int cbase)
{
    constexpr int KST = 34;                       // padded K-stride (bank spread)
    __shared__ __attribute__((aligned(16))) __bf16 w1s[768 * KST];
    const int tid  = threadIdx.x;
    const int wv   = tid >> 5;
    const int lane = tid & 31;
    const int lm   = lane & 15;
    const int mh   = (lane < 16) ? 0 : 8;
    const int r0   = blockIdx.x * 32;             // flattened row base (b*512+i)
    const int cl   = blockIdx.y;
    const float* w1c = W1 + (size_t)(cbase + cl) * 768 * 768;

    const v8f vz = {0.f, 0.f, 0.f, 0.f, 0.f, 0.f, 0.f, 0.f};
    v8f acc[2][6];
    #pragma unroll
    for (int a = 0; a < 2; ++a)
        #pragma unroll
        for (int e = 0; e < 6; ++e) acc[a][e] = vz;

    for (int k0 = 0; k0 < 768; k0 += 32) {
        v16bf Ah[2], Al[2];
        #pragma unroll
        for (int it = 0; it < 2; ++it) {
            Ah[it] = load_afrag(Hshi + (size_t)(r0 + it * 16) * 768 + k0, 768, lane);
            Al[it] = load_afrag(Hslo + (size_t)(r0 + it * 16) * 768 + k0, 768, lane);
        }
        // stage hi-plane of W1 chunk, transposed to [e][kk]
        __syncthreads();
        for (int idx = tid; idx < 6144; idx += 256) {
            int e4 = idx % 192, kk = idx / 192;
            f32x4 v4 = *(const f32x4*)(w1c + (size_t)(k0 + kk) * 768 + e4 * 4);
            #pragma unroll
            for (int q = 0; q < 4; ++q)
                w1s[(e4 * 4 + q) * KST + kk] = (__bf16)v4[q];
        }
        __syncthreads();
        #pragma unroll
        for (int et = 0; et < 6; ++et) {
            v16bf Bh = load_bfrag(&w1s[(size_t)((wv * 6 + et) * 16) * KST], KST, lane);
            #pragma unroll
            for (int it = 0; it < 2; ++it) {
                acc[it][et] = wmma_acc(Ah[it], Bh, acc[it][et]);
                acc[it][et] = wmma_acc(Al[it], Bh, acc[it][et]);
            }
        }
        __syncthreads();
        // stage lo-plane
        for (int idx = tid; idx < 6144; idx += 256) {
            int e4 = idx % 192, kk = idx / 192;
            f32x4 v4 = *(const f32x4*)(w1c + (size_t)(k0 + kk) * 768 + e4 * 4);
            #pragma unroll
            for (int q = 0; q < 4; ++q) {
                float x = v4[q];
                __bf16 h = (__bf16)x;
                w1s[(e4 * 4 + q) * KST + kk] = (__bf16)(x - (float)h);
            }
        }
        __syncthreads();
        #pragma unroll
        for (int et = 0; et < 6; ++et) {
            v16bf Bl = load_bfrag(&w1s[(size_t)((wv * 6 + et) * 16) * KST], KST, lane);
            #pragma unroll
            for (int it = 0; it < 2; ++it)
                acc[it][et] = wmma_acc(Ah[it], Bl, acc[it][et]);
        }
    }
    // store T (bf16 hi/lo): C/D layout -> lane col e = lm, rows v + mh
    const int b  = r0 >> 9;
    const int i0 = r0 & 511;
    #pragma unroll
    for (int it = 0; it < 2; ++it)
        #pragma unroll
        for (int et = 0; et < 6; ++et) {
            const int e = (wv * 6 + et) * 16 + lm;
            #pragma unroll
            for (int v = 0; v < 8; ++v) {
                const int i = i0 + it * 16 + mh + v;
                size_t idx = ((size_t)(b * NCH + cl) * 512 + i) * 768 + e;
                float x = acc[it][et][v];
                __bf16 h = (__bf16)x;
                Thi[idx] = h;
                Tlo[idx] = (__bf16)(x - (float)h);
            }
        }
}

// ---------------------------------------------------------------------------
// Pass 2: S[b,i,j,c] = sum_e T[b,c,i,e]*He[b,j,e] + Ls[b,i,c]+Le[b,j,c]+W2b[c]+bias[c]
// grid = (4 j-chunks, 32 i-tiles, b * (NCH/8) c-groups).  One c per wave,
// 8 j-tile accumulators; He planes serve B fragments directly (N-major in j).
// Output staged through LDS so the c-fastest layout is written in 32B runs.
// ---------------------------------------------------------------------------
__global__ void __launch_bounds__(256) pass2_S_kernel(
    const __bf16* __restrict__ Thi, const __bf16* __restrict__ Tlo,
    const __bf16* __restrict__ Hehi, const __bf16* __restrict__ Helo,
    const float* __restrict__ Ls, const float* __restrict__ Le,
    const float* __restrict__ W2b, const float* __restrict__ biasp,
    float* __restrict__ out, int NCH, int cbase)
{
    __shared__ float st[16 * 64 * 9];
    const int tid  = threadIdx.x;
    const int wv   = tid >> 5;
    const int lane = tid & 31;
    const int lm   = lane & 15;
    const int mh   = (lane < 16) ? 0 : 8;
    const int j0   = blockIdx.x * 128;
    const int i0   = blockIdx.y * 16;
    const int ngc  = NCH >> 3;
    const int b    = blockIdx.z / ngc;
    const int cg   = blockIdx.z % ngc;
    const int cl   = cg * 8 + wv;
    const int cglob = cbase + cg * 8;

    const v8f vz = {0.f, 0.f, 0.f, 0.f, 0.f, 0.f, 0.f, 0.f};
    v8f acc[8];
    #pragma unroll
    for (int jt = 0; jt < 8; ++jt) acc[jt] = vz;

    const size_t tbase = ((size_t)(b * NCH + cl) * 512 + i0) * 768;
    for (int k0 = 0; k0 < 768; k0 += 32) {
        v16bf Ah = load_afrag(Thi + tbase + k0, 768, lane);
        v16bf Al = load_afrag(Tlo + tbase + k0, 768, lane);
        #pragma unroll
        for (int jt = 0; jt < 8; ++jt) {
            const size_t hb = (size_t)(b * 512 + j0 + jt * 16) * 768 + k0;
            v16bf Bh = load_bfrag(Hehi + hb, 768, lane);
            v16bf Bl = load_bfrag(Helo + hb, 768, lane);
            acc[jt] = wmma_acc(Ah, Bh, acc[jt]);
            acc[jt] = wmma_acc(Al, Bh, acc[jt]);
            acc[jt] = wmma_acc(Ah, Bl, acc[jt]);
        }
    }
    // epilogue: stage 16 x 64 x 8c tiles in LDS, add linear terms, write out
    #pragma unroll
    for (int phase = 0; phase < 2; ++phase) {
        __syncthreads();
        #pragma unroll
        for (int jt4 = 0; jt4 < 4; ++jt4) {
            const int jt = phase * 4 + jt4;
            const int jj = jt * 16 + lm - phase * 64;
            #pragma unroll
            for (int v = 0; v < 8; ++v)
                st[((mh + v) * 64 + jj) * 9 + wv] = acc[jt][v];
        }
        __syncthreads();
        #pragma unroll
        for (int r = 0; r < 4; ++r) {
            const int p  = tid + 256 * r;
            const int ii = p >> 6;
            const int jj = p & 63;
            const int i  = i0 + ii;
            const int j  = j0 + phase * 64 + jj;
            const float* lsrow = Ls + (size_t)(b * 512 + i) * 256 + cglob;
            const float* lerow = Le + (size_t)(b * 512 + j) * 256 + cglob;
            float o[8];
            #pragma unroll
            for (int cc = 0; cc < 8; ++cc)
                o[cc] = st[(ii * 64 + jj) * 9 + cc] + lsrow[cc] + lerow[cc]
                        + W2b[cglob + cc] + biasp[cglob + cc];
            float* po = out + ((size_t)(b * 512 + i) * 512 + j) * 256 + cglob;
            f32x4 w0, w1;
            #pragma unroll
            for (int cc = 0; cc < 4; ++cc) { w0[cc] = o[cc]; w1[cc] = o[cc + 4]; }
            *(f32x4*)(po)     = w0;
            *(f32x4*)(po + 4) = w1;
        }
    }
}

// ---------------------------------------------------------------------------
extern "C" void kernel_launch(void* const* d_in, const int* in_sizes, int n_in,
                              void* d_out, int out_size, void* d_ws, size_t ws_size,
                              hipStream_t stream)
{
    (void)in_sizes; (void)n_in; (void)out_size;
    const float* hidden = (const float*)d_in[0];
    const float* sw1 = (const float*)d_in[1];
    const float* sb1 = (const float*)d_in[2];
    const float* sw2 = (const float*)d_in[3];
    const float* sb2 = (const float*)d_in[4];
    const float* ew1 = (const float*)d_in[5];
    const float* eb1 = (const float*)d_in[6];
    const float* ew2 = (const float*)d_in[7];
    const float* eb2 = (const float*)d_in[8];
    const float* W1  = (const float*)d_in[9];
    const float* W2w = (const float*)d_in[10];
    const float* W2b = (const float*)d_in[11];
    const float* bias = (const float*)d_in[12];
    float* out = (float*)d_out;

    const int M = 1024, D = 768, C = 256;

    char* ws = (char*)d_ws;
    size_t off = 0;
    auto alloc = [&](size_t bytes) -> char* {
        char* p = ws + off;
        off = (off + bytes + 255) & ~(size_t)255;
        return p;
    };

    float*  h     = (float*) alloc((size_t)M * D * 4);
    float*  Hs    = (float*) alloc((size_t)M * D * 4);
    float*  He    = (float*) alloc((size_t)M * D * 4);
    float*  Lsb   = (float*) alloc((size_t)M * C * 4);
    float*  Leb   = (float*) alloc((size_t)M * C * 4);
    __bf16* Hshi  = (__bf16*)alloc((size_t)M * D * 2);
    __bf16* Hslo  = (__bf16*)alloc((size_t)M * D * 2);
    __bf16* Hehi  = (__bf16*)alloc((size_t)M * D * 2);
    __bf16* Helo  = (__bf16*)alloc((size_t)M * D * 2);

    // pick channel chunk to fit T hi/lo planes in remaining workspace
    const size_t perchan = (size_t)2 * 512 * 768 * 2 * 2;   // 3 MiB / channel
    int NCH = 8;
    {
        size_t rem = (ws_size > off) ? (ws_size - off) : 0;
        const int cands[6] = {256, 128, 64, 32, 16, 8};
        for (int ci = 0; ci < 6; ++ci)
            if ((size_t)cands[ci] * perchan <= rem) { NCH = cands[ci]; break; }
    }
    __bf16* Thi = (__bf16*)alloc((size_t)2 * NCH * 512 * 768 * 2);
    __bf16* Tlo = (__bf16*)alloc((size_t)2 * NCH * 512 * 768 * 2);

    dim3 blk(256);
    // MLPs (f32)
    gemm_xwt_kernel<<<dim3(16, 12), blk, 0, stream>>>(hidden, sw1, sb1, h,  M, D, D, D, 0, 1);
    gemm_xwt_kernel<<<dim3(16, 12), blk, 0, stream>>>(h,      sw2, sb2, Hs, M, D, D, D, 0, 0);
    gemm_xwt_kernel<<<dim3(16, 12), blk, 0, stream>>>(hidden, ew1, eb1, h,  M, D, D, D, 0, 1);
    gemm_xwt_kernel<<<dim3(16, 12), blk, 0, stream>>>(h,      ew2, eb2, He, M, D, D, D, 0, 0);
    // linear heads Ls = Hs @ Ws^T, Le = He @ We^T (W2_w is (C, 2D))
    gemm_xwt_kernel<<<dim3(16, 4), blk, 0, stream>>>(Hs, W2w, nullptr, Lsb, M, C, D, 2 * D, 0, 0);
    gemm_xwt_kernel<<<dim3(16, 4), blk, 0, stream>>>(He, W2w, nullptr, Leb, M, C, D, 2 * D, D, 0);
    // bf16 hi/lo planes
    split_kernel<<<dim3((M * D + 255) / 256), blk, 0, stream>>>(Hs, Hshi, Hslo, M * D);
    split_kernel<<<dim3((M * D + 255) / 256), blk, 0, stream>>>(He, Hehi, Helo, M * D);

    // bilinear core, chunked over channels
    for (int cb = 0; cb < C; cb += NCH) {
        pass1_T_kernel<<<dim3(32, NCH), blk, 0, stream>>>(Hshi, Hslo, W1, Thi, Tlo, NCH, cb);
        pass2_S_kernel<<<dim3(4, 32, 2 * (NCH / 8)), blk, 0, stream>>>(
            Thi, Tlo, Hehi, Helo, Lsb, Leb, W2b, bias, out, NCH, cb);
    }
}
